// KANActorCritic_27771258536159
// MI455X (gfx1250) — compile-verified, matrix operand order checked by reading, not run on hardware
//
#include <hip/hip_runtime.h>

typedef __attribute__((ext_vector_type(16))) __bf16 v16bf;
typedef __attribute__((ext_vector_type(8)))  __bf16 v8bf;
typedef __attribute__((ext_vector_type(8)))  float  v8f;

#define BB   512
#define LL   128
#define FDIM 8
#define HH   512
#define MM   (BB * LL)   // 65536 rows

#if defined(__gfx1250__) && \
    __has_builtin(__builtin_amdgcn_global_load_async_to_lds_b128) && \
    __has_builtin(__builtin_amdgcn_s_wait_asynccnt)
#define USE_ASYNC_LDS 1
#else
#define USE_ASYNC_LDS 0
#endif

#if USE_ASYNC_LDS
typedef int i32x4 __attribute__((vector_size(16)));
typedef __attribute__((address_space(1))) i32x4 g_i32x4;   // global-side 16B vector
typedef __attribute__((address_space(3))) i32x4 l_i32x4;   // LDS-side 16B vector
#endif

__device__ __forceinline__ __bf16 f2bf(float f) {
  union { float f; unsigned u; } in; in.f = f;
  unsigned u = in.u;
  unsigned r = u + 0x7FFFu + ((u >> 16) & 1u);   // round-to-nearest-even
  union { unsigned short s; __bf16 b; } out; out.s = (unsigned short)(r >> 16);
  return out.b;
}

// async (CDNA5) or fallback synchronous 16B global->LDS copy
__device__ __forceinline__ void cp16_to_lds(const __bf16* g, __bf16* l) {
#if USE_ASYNC_LDS
  __builtin_amdgcn_global_load_async_to_lds_b128(
      (g_i32x4*)(__bf16*)g, (l_i32x4*)l, 0, 0);
#else
  *(uint4*)l = *(const uint4*)g;
#endif
}
__device__ __forceinline__ void cp_wait_all() {
#if USE_ASYNC_LDS
  __builtin_amdgcn_s_wait_asynccnt(0);
#endif
}

// ---------------- input projection: h = x @ w0 + b0 (K=8) ----------------
__global__ __launch_bounds__(256) void k_inproj(
    const float* __restrict__ x, const float* __restrict__ w0,
    const float* __restrict__ b0, float* __restrict__ h32,
    __bf16* __restrict__ hbf) {
  size_t gid = (size_t)blockIdx.x * blockDim.x + threadIdx.x;
  size_t row = gid >> 9;
  int col = (int)(gid & 511);
  const float* xr = x + row * FDIM;
  float acc = b0[col];
#pragma unroll
  for (int f = 0; f < FDIM; ++f) acc += xr[f] * w0[f * HH + col];
  h32[gid] = acc;
  hbf[gid] = f2bf(acc);
}

// ---------------- weight transpose+convert: f32[K,N] -> bf16[N,K] --------
__global__ void k_wprep(const float* __restrict__ src, __bf16* __restrict__ dst,
                        int K, int N) {
  int idx = blockIdx.x * blockDim.x + threadIdx.x;
  if (idx >= K * N) return;
  int k = idx / N, n = idx - k * N;
  dst[(size_t)n * K + k] = f2bf(src[idx]);
}

// ---------------- bf16 WMMA GEMM: D[M,N] = A[M,K] @ Bt[N,K]^T + bias -----
// 256 threads (8 waves), 128x128 block tile, K-step 32, double-buffered LDS
// filled via GLOBAL_LOAD_ASYNC_TO_LDS_B128 (overlaps next fill with WMMA).
__global__ __launch_bounds__(256) void k_gemm_bf16(
    const __bf16* __restrict__ A, const __bf16* __restrict__ Bt,
    const float* __restrict__ bias, float* __restrict__ D, int K, int N) {
  __shared__ __align__(16) __bf16 As[2][128 * 40];  // 40-elem pitch: conflict-free frags
  __shared__ __align__(16) __bf16 Bs[2][128 * 40];

  const int tid   = threadIdx.x;
  const int lane  = tid & 31;
  const int wid   = tid >> 5;
  const int waveM = wid >> 2;            // 0..1
  const int waveN = wid & 3;             // 0..3
  const int mBlk  = blockIdx.x * 128;
  const int nBlk  = blockIdx.y * 128;

  const int lr0 = tid >> 2;              // rows lr0, lr0+64
  const int lc  = tid & 3;               // 16B chunk within 64B row

  const __bf16* gA0 = A  + (size_t)(mBlk + lr0)      * K + lc * 8;
  const __bf16* gA1 = A  + (size_t)(mBlk + lr0 + 64) * K + lc * 8;
  const __bf16* gB0 = Bt + (size_t)(nBlk + lr0)      * K + lc * 8;
  const __bf16* gB1 = Bt + (size_t)(nBlk + lr0 + 64) * K + lc * 8;
  const int lOff0 = lr0 * 40 + lc * 8;
  const int lOff1 = (lr0 + 64) * 40 + lc * 8;

  const v8f vzero = {0.f, 0.f, 0.f, 0.f, 0.f, 0.f, 0.f, 0.f};
  v8f acc[4][2];
#pragma unroll
  for (int i = 0; i < 4; ++i)
#pragma unroll
    for (int j = 0; j < 2; ++j) acc[i][j] = vzero;

  const int nIter = K >> 5;

  // prologue: fill buffer 0
  cp16_to_lds(gA0, &As[0][lOff0]);
  cp16_to_lds(gA1, &As[0][lOff1]);
  cp16_to_lds(gB0, &Bs[0][lOff0]);
  cp16_to_lds(gB1, &Bs[0][lOff1]);

  const int klo = (lane >> 4) * 8;       // lanes 0-15: K 0..7/16..23; 16-31: 8..15/24..31
  const int ml  = lane & 15;

  for (int it = 0; it < nIter; ++it) {
    const int buf = it & 1;
    cp_wait_all();                       // my wave's async fills for `buf` done
    __syncthreads();                     // all waves' fills visible; prev reads of buf^1 done
    if (it + 1 < nIter) {
      const int k0 = (it + 1) << 5;
      cp16_to_lds(gA0 + k0, &As[buf ^ 1][lOff0]);
      cp16_to_lds(gA1 + k0, &As[buf ^ 1][lOff1]);
      cp16_to_lds(gB0 + k0, &Bs[buf ^ 1][lOff0]);
      cp16_to_lds(gB1 + k0, &Bs[buf ^ 1][lOff1]);
    }

    v16bf afrag[4];
#pragma unroll
    for (int i = 0; i < 4; ++i) {
      int mrow = waveM * 64 + i * 16 + ml;
      v8bf lo = *(const v8bf*)&As[buf][mrow * 40 + klo];
      v8bf hi = *(const v8bf*)&As[buf][mrow * 40 + 16 + klo];
      afrag[i] = __builtin_shufflevector(lo, hi, 0,1,2,3,4,5,6,7,8,9,10,11,12,13,14,15);
    }
    v16bf bfrag[2];
#pragma unroll
    for (int j = 0; j < 2; ++j) {
      int nrow = waveN * 32 + j * 16 + ml;
      v8bf lo = *(const v8bf*)&Bs[buf][nrow * 40 + klo];
      v8bf hi = *(const v8bf*)&Bs[buf][nrow * 40 + 16 + klo];
      bfrag[j] = __builtin_shufflevector(lo, hi, 0,1,2,3,4,5,6,7,8,9,10,11,12,13,14,15);
    }
#pragma unroll
    for (int i = 0; i < 4; ++i)
#pragma unroll
      for (int j = 0; j < 2; ++j)
        acc[i][j] = __builtin_amdgcn_wmma_f32_16x16x32_bf16(
            false, afrag[i], false, bfrag[j], (short)0, acc[i][j], false, false);

    __syncthreads();                     // done reading `buf` before it is refilled
  }

  const int mh = (lane >> 4) * 8;        // C/D layout: VGPR v -> M=v (lanes<16) / v+8
#pragma unroll
  for (int i = 0; i < 4; ++i)
#pragma unroll
    for (int j = 0; j < 2; ++j) {
      int col = nBlk + waveN * 32 + j * 16 + ml;
      float bv = bias[col];
#pragma unroll
      for (int v = 0; v < 8; ++v) {
        int row = mBlk + waveM * 64 + i * 16 + mh + v;
        D[(size_t)row * N + col] = acc[i][j][v] + bv;
      }
    }
}

// ---------------- depthwise conv3 + silu + dt bottleneck + gate ----------
__global__ __launch_bounds__(256) void k_conv_dt(
    const float* __restrict__ zi, const float* __restrict__ conv_w,
    const float* __restrict__ conv_b, const float* __restrict__ xp_w,
    const float* __restrict__ xp_b, const float* __restrict__ dt_w,
    const float* __restrict__ dt_b, __bf16* __restrict__ catbf) {
  __shared__ float red[256 * 16];
  __shared__ float tmpv[16];

  const int row = blockIdx.x;
  const int l   = row & (LL - 1);
  const int t   = threadIdx.x;
  const float* zrow = zi + (size_t)row * (2 * HH);

  float myxc[2];
#pragma unroll
  for (int q = 0; q < 2; ++q) {
    int h = t + q * 256;
    float xm1 = (l > 0)      ? zi[((size_t)row - 1) * (2 * HH) + HH + h] : 0.f;
    float x0  = zrow[HH + h];
    float xp1 = (l < LL - 1) ? zi[((size_t)row + 1) * (2 * HH) + HH + h] : 0.f;
    float c = xm1 * conv_w[h * 3 + 0] + x0 * conv_w[h * 3 + 1] +
              xp1 * conv_w[h * 3 + 2] + conv_b[h];
    c = c / (1.f + __expf(-c));          // silu
    myxc[q] = c;
  }
  float p[16];
#pragma unroll
  for (int j = 0; j < 16; ++j) p[j] = 0.f;
#pragma unroll
  for (int q = 0; q < 2; ++q) {
    int h = t + q * 256;
    float v = myxc[q];
#pragma unroll
    for (int j = 0; j < 16; ++j) p[j] += v * xp_w[h * 16 + j];
  }
#pragma unroll
  for (int j = 0; j < 16; ++j) red[t * 16 + j] = p[j];
  __syncthreads();
  for (int s = 128; s > 0; s >>= 1) {
    if (t < s)
      for (int j = 0; j < 16; ++j) red[t * 16 + j] += red[(t + s) * 16 + j];
    __syncthreads();
  }
  if (t < 16) tmpv[t] = red[t] + xp_b[t];
  __syncthreads();

#pragma unroll
  for (int q = 0; q < 2; ++q) {
    int h = t + q * 256;
    float acc = dt_b[h];
#pragma unroll
    for (int j = 0; j < 16; ++j) acc += tmpv[j] * dt_w[j * HH + h];
    float sp = (acc > 20.f) ? acc : log1pf(__expf(acc));   // softplus
    float zv = zrow[h];
    float sz = zv / (1.f + __expf(-zv));
    catbf[(size_t)row * (2 * HH) + h]      = f2bf(myxc[q] * sp);
    catbf[(size_t)row * (2 * HH) + HH + h] = f2bf(sz);
  }
}

// ---------------- residual add + layernorm -------------------------------
__global__ __launch_bounds__(256) void k_add_ln(
    const float* __restrict__ y, float* __restrict__ h32,
    const float* __restrict__ g, const float* __restrict__ b,
    __bf16* __restrict__ hbf) {
  __shared__ float red[256];
  const int row = blockIdx.x;
  const int t = threadIdx.x;
  size_t base = (size_t)row * HH;
  float v0 = y[base + t]       + h32[base + t];
  float v1 = y[base + t + 256] + h32[base + t + 256];
  red[t] = v0 + v1;
  __syncthreads();
  for (int s = 128; s > 0; s >>= 1) { if (t < s) red[t] += red[t + s]; __syncthreads(); }
  float mu = red[0] * (1.f / HH);
  __syncthreads();
  float d0 = v0 - mu, d1 = v1 - mu;
  red[t] = d0 * d0 + d1 * d1;
  __syncthreads();
  for (int s = 128; s > 0; s >>= 1) { if (t < s) red[t] += red[t + s]; __syncthreads(); }
  float rstd = rsqrtf(red[0] * (1.f / HH) + 1e-5f);
  float o0 = d0 * rstd * g[t]       + b[t];
  float o1 = d1 * rstd * g[t + 256] + b[t + 256];
  h32[base + t] = o0;       h32[base + t + 256] = o1;
  hbf[base + t] = f2bf(o0); hbf[base + t + 256] = f2bf(o1);
}

// ---------------- gating + actor/critic head -----------------------------
__global__ __launch_bounds__(128) void k_head(
    const float* __restrict__ h32, const float* __restrict__ gumbel,
    const float* __restrict__ gate_w, const float* __restrict__ gate_b,
    const float* __restrict__ actor_w, const float* __restrict__ actor_b,
    const float* __restrict__ critic_w, const float* __restrict__ critic_b,
    float* __restrict__ out) {
  __shared__ float sreg[HH];
  __shared__ float sdec[HH];
  __shared__ float red[128 * 4];
  __shared__ int sel;
  const int bidx = blockIdx.x;
  const int t = threadIdx.x;

#pragma unroll
  for (int q = 0; q < 4; ++q) {
    int h = t + q * 128;
    float acc = 0.f;
    for (int l = 0; l < LL; ++l)
      acc += h32[((size_t)bidx * LL + l) * HH + h];
    sreg[h] = acc * (1.f / LL);
    sdec[h] = h32[((size_t)bidx * LL + (LL - 1)) * HH + h];
  }
  __syncthreads();

  float p[4] = {0.f, 0.f, 0.f, 0.f};
#pragma unroll
  for (int q = 0; q < 4; ++q) {
    int h = t + q * 128;
    float r = sreg[h];
#pragma unroll
    for (int e = 0; e < 4; ++e) p[e] += r * gate_w[h * 4 + e];
  }
#pragma unroll
  for (int e = 0; e < 4; ++e) red[t * 4 + e] = p[e];
  __syncthreads();
  for (int s = 64; s > 0; s >>= 1) {
    if (t < s) for (int e = 0; e < 4; ++e) red[t * 4 + e] += red[(t + s) * 4 + e];
    __syncthreads();
  }
  if (t == 0) {
    float best = -1e30f; int bi = 0;
    for (int e = 0; e < 4; ++e) {
      float gl = red[e] + gate_b[e] + gumbel[bidx * 4 + e];
      if (gl > best) { best = gl; bi = e; }   // first-max tie-break like argmax
    }
    sel = bi;
  }
  __syncthreads();
  int estar = sel;
  // gate = hard + y - stop_grad(y) == one-hot numerically in forward pass
  if (t < 4) out[BB * 3 + BB + bidx * 4 + t] = (t == estar) ? 1.f : 0.f;

  float pa[4] = {0.f, 0.f, 0.f, 0.f};
#pragma unroll
  for (int q = 0; q < 4; ++q) {
    int h = t + q * 128;
    float d = sdec[h];
#pragma unroll
    for (int a = 0; a < 3; ++a) pa[a] += d * actor_w[((size_t)estar * HH + h) * 3 + a];
    pa[3] += d * critic_w[(size_t)estar * HH + h];
  }
  __syncthreads();
#pragma unroll
  for (int e = 0; e < 4; ++e) red[t * 4 + e] = pa[e];
  __syncthreads();
  for (int s = 64; s > 0; s >>= 1) {
    if (t < s) for (int e = 0; e < 4; ++e) red[t * 4 + e] += red[(t + s) * 4 + e];
    __syncthreads();
  }
  if (t == 0) {
    for (int a = 0; a < 3; ++a) out[bidx * 3 + a] = red[a] + actor_b[estar * 3 + a];
    out[BB * 3 + bidx] = red[3] + critic_b[estar];
  }
}

extern "C" void kernel_launch(void* const* d_in, const int* in_sizes, int n_in,
                              void* d_out, int out_size, void* d_ws, size_t ws_size,
                              hipStream_t stream) {
  (void)in_sizes; (void)n_in; (void)out_size; (void)ws_size;
  const float* x        = (const float*)d_in[0];
  const float* gumbel   = (const float*)d_in[1];
  const float* w0       = (const float*)d_in[2];
  const float* b0       = (const float*)d_in[3];
  const float* ln_g     = (const float*)d_in[4];
  const float* ln_b     = (const float*)d_in[5];
  const float* gate_w   = (const float*)d_in[6];
  const float* gate_b   = (const float*)d_in[7];
  const float* actor_w  = (const float*)d_in[8];
  const float* actor_b  = (const float*)d_in[9];
  const float* critic_w = (const float*)d_in[10];
  const float* critic_b = (const float*)d_in[11];

  const float* m_in_w[2]   = { (const float*)d_in[12], (const float*)d_in[22] };
  const float* m_in_b[2]   = { (const float*)d_in[13], (const float*)d_in[23] };
  const float* m_conv_w[2] = { (const float*)d_in[14], (const float*)d_in[24] };
  const float* m_conv_b[2] = { (const float*)d_in[15], (const float*)d_in[25] };
  const float* m_xp_w[2]   = { (const float*)d_in[16], (const float*)d_in[26] };
  const float* m_xp_b[2]   = { (const float*)d_in[17], (const float*)d_in[27] };
  const float* m_dt_w[2]   = { (const float*)d_in[18], (const float*)d_in[28] };
  const float* m_dt_b[2]   = { (const float*)d_in[19], (const float*)d_in[29] };
  const float* m_out_w[2]  = { (const float*)d_in[20], (const float*)d_in[30] };
  const float* m_out_b[2]  = { (const float*)d_in[21], (const float*)d_in[31] };

  char* p = (char*)d_ws;
  auto carve = [&](size_t bytes) {
    char* r = p;
    p += (bytes + 255) & ~(size_t)255;
    return r;
  };
  float*  h32   = (float*)  carve((size_t)MM * HH * 4);
  __bf16* hbf   = (__bf16*) carve((size_t)MM * HH * 2);
  float*  zi32  = (float*)  carve((size_t)MM * 2 * HH * 4);
  __bf16* catbf = (__bf16*) carve((size_t)MM * 2 * HH * 2);
  __bf16* wInT[2]  = { (__bf16*)carve((size_t)HH * 2 * HH * 2),
                       (__bf16*)carve((size_t)HH * 2 * HH * 2) };
  __bf16* wOutT[2] = { (__bf16*)carve((size_t)HH * 2 * HH * 2),
                       (__bf16*)carve((size_t)HH * 2 * HH * 2) };
  float* out32 = zi32;   // zi is dead by the time GEMM2 writes; safe alias

  for (int m = 0; m < 2; ++m) {
    k_wprep<<<(HH * 2 * HH + 255) / 256, 256, 0, stream>>>(m_in_w[m],  wInT[m],  HH,     2 * HH);
    k_wprep<<<(HH * 2 * HH + 255) / 256, 256, 0, stream>>>(m_out_w[m], wOutT[m], 2 * HH, HH);
  }
  k_inproj<<<(MM * HH) / 256, 256, 0, stream>>>(x, w0, b0, h32, hbf);

  for (int m = 0; m < 2; ++m) {
    dim3 g1(MM / 128, (2 * HH) / 128);
    k_gemm_bf16<<<g1, 256, 0, stream>>>(hbf, wInT[m], m_in_b[m], zi32, HH, 2 * HH);
    k_conv_dt<<<MM, 256, 0, stream>>>(zi32, m_conv_w[m], m_conv_b[m], m_xp_w[m],
                                      m_xp_b[m], m_dt_w[m], m_dt_b[m], catbf);
    dim3 g2(MM / 128, HH / 128);
    k_gemm_bf16<<<g2, 256, 0, stream>>>(catbf, wOutT[m], m_out_b[m], out32, 2 * HH, HH);
    k_add_ln<<<MM, 256, 0, stream>>>(out32, h32, ln_g, ln_b, hbf);
  }
  k_head<<<BB, 128, 0, stream>>>(h32, gumbel, gate_w, gate_b, actor_w, actor_b,
                                 critic_w, critic_b, (float*)d_out);
}
